// TurboQuantizer_68539088110214
// MI455X (gfx1250) — matrix operand order repaired and checked by reading, not compiled, not run on hardware
//
#include <hip/hip_runtime.h>

typedef __attribute__((ext_vector_type(16))) _Float16 v16h;
typedef __attribute__((ext_vector_type(8)))  _Float16 v8h;
typedef __attribute__((ext_vector_type(8)))  float    v8f;

constexpr int DIM    = 128;          // vector dimension
constexpr int PITCH  = 136;          // padded LDS row pitch (halfs) -> bank-conflict-free
constexpr int MATH   = DIM * PITCH;  // halfs per staged 128x128 matrix
constexpr int STAGEH = 16 * PITCH;   // halfs per 16-row stage buffer
constexpr int WAVES  = 4;
constexpr int ROWS_PER_BLOCK = WAVES * 16;
constexpr int GRID_BLOCKS = 1024;    // persistent-style blocks; amortize matrix staging

__device__ __forceinline__ v16h combine8(v8h lo, v8h hi) {
  v16h r;
#pragma unroll
  for (int j = 0; j < 8; ++j) { r[j] = lo[j]; r[j + 8] = hi[j]; }
  return r;
}

// A fragment for V_WMMA_F32_16X16X32_F16.
// lane<16 (kh=0): K = 32t+{0..7}, {16..23}; lane>=16 (kh=1): K = 32t+{8..15}, {24..31}
__device__ __forceinline__ v16h load_a_frag(const _Float16* rowp, int t, int kh) {
  v8h lo = *(const v8h*)(rowp + 32 * t + 8 * kh);
  v8h hi = *(const v8h*)(rowp + 32 * t + 16 + 8 * kh);
  return combine8(lo, hi);
}

// B fragment: matrices stored "column of B"-major: mat[n*PITCH + k] = B[k][n].
// lane n=(lane&15)+16*nt holds K = 32t + 8*(lane>>4) + {0..7}, {16..23}
__device__ __forceinline__ v16h load_b_frag(const _Float16* mat, int lane, int nt, int t) {
  const int n = (nt << 4) + (lane & 15);
  const int k = (t << 5) + ((lane >> 4) << 3);
  const _Float16* p = mat + n * PITCH + k;
  v8h lo = *(const v8h*)p;
  v8h hi = *(const v8h*)(p + 16);
  return combine8(lo, hi);
}

// (16x128) @ (128x128) -> 16x128, accumulated into 8 C-tiles
__device__ __forceinline__ void gemm_16x128(const v16h a[4], const _Float16* mat,
                                            int lane, v8f acc[8]) {
#pragma unroll
  for (int nt = 0; nt < 8; ++nt) {
#pragma unroll
    for (int t = 0; t < 4; ++t) {
      v16h b = load_b_frag(mat, lane, nt, t);
      acc[nt] = __builtin_amdgcn_wmma_f32_16x16x32_f16(
          false, a[t], false, b, (short)0, acc[nt], false, false);
    }
  }
}

__global__ __launch_bounds__(WAVES * 32)
void turboquant_roundtrip_kernel(const float* __restrict__ x,
                                 const float* __restrict__ Pi,
                                 const float* __restrict__ S,
                                 const float* __restrict__ cent,
                                 float* __restrict__ out, int nTiles) {
  extern __shared__ _Float16 smem[];
  _Float16* mPiT = smem;             // B for x_hat @ Pi^T : [n][k] = Pi[n][k]
  _Float16* mPi  = smem + 1 * MATH;  // B for recon @ Pi   : [n][k] = Pi[k][n]
  _Float16* mST  = smem + 2 * MATH;  // B for resid @ S^T  : [n][k] = S[n][k]
  _Float16* mS   = smem + 3 * MATH;  // B for signs @ S    : [n][k] = S[k][n]
  _Float16* stages = smem + 4 * MATH;
  float* snorm = (float*)(smem + 4 * MATH + WAVES * 2 * STAGEH);

  const int tid  = threadIdx.x;
  const int wv   = tid >> 5;
  const int lane = tid & 31;
  const int ln   = lane & 15;  // column / row selector within 16
  const int hh   = lane >> 4;  // half selector

  // ---- stage rotation matrices into LDS as f16, both majors (once per block) ----
  for (int i = tid; i < DIM * DIM; i += blockDim.x) {
    const int r = i >> 7, c = i & (DIM - 1);
    const float pv = Pi[i], sv = S[i];
    mPiT[r * PITCH + c] = (_Float16)pv;
    mPi [c * PITCH + r] = (_Float16)pv;
    mST [r * PITCH + c] = (_Float16)sv;
    mS  [c * PITCH + r] = (_Float16)sv;
  }
  const float c0 = cent[0], c1 = cent[1], c2 = cent[2], c3 = cent[3];
  const float t01 = 0.5f * (c0 + c1), t12 = 0.5f * (c1 + c2), t23 = 0.5f * (c2 + c3);
  __syncthreads();  // only block-wide barrier needed: stage buffers below are per-wave

  _Float16* st0 = stages + wv * (2 * STAGEH);  // x_hat (f16)
  _Float16* st1 = st0 + STAGEH;                // recon / residual / signs (f16)
  const _Float16* arow0 = st0 + ln * PITCH;
  const _Float16* arow1 = st1 + ln * PITCH;
  const float CORR = 1.2533141373155003f / 128.0f;  // sqrt(pi/2)/m

  // ---- grid-stride loop over 64-row tiles ----
  for (int tile = blockIdx.x; tile < nTiles; tile += gridDim.x) {
    const int rowBase = tile * ROWS_PER_BLOCK + wv * 16;

    // phase A: load 16x128 f32 tile, per-row norm, stage x_hat (f16)
    {
      const float4* rp = (const float4*)(x + (size_t)(rowBase + ln) * DIM + hh * 64);
      float4 v[16];
      float ss = 0.f;
#pragma unroll
      for (int i = 0; i < 16; ++i) {
        v[i] = rp[i];
        ss += v[i].x * v[i].x + v[i].y * v[i].y + v[i].z * v[i].z + v[i].w * v[i].w;
      }
      // prefetch next tile's rows into cache while we crunch this one
      const int nextTile = tile + gridDim.x;
      if (nextTile < nTiles) {
        const char* pf = (const char*)(x + (size_t)(nextTile * ROWS_PER_BLOCK +
                                                    wv * 16 + ln) * DIM + hh * 64);
        __builtin_prefetch(pf, 0, 1);        // -> global_prefetch_b8
        __builtin_prefetch(pf + 128, 0, 1);
      }
      ss += __shfl_xor(ss, 16, 32);  // combine the two half-row partials
      const float nrm = sqrtf(ss);
      const float inv = 1.f / (nrm + 1e-8f);
      if (hh == 0) snorm[wv * 16 + ln] = (float)(_Float16)nrm;  // fp16 round-trip
      _Float16* w = st0 + ln * PITCH + hh * 64;
#pragma unroll
      for (int i = 0; i < 16; ++i) {
        w[4 * i + 0] = (_Float16)(v[i].x * inv);
        w[4 * i + 1] = (_Float16)(v[i].y * inv);
        w[4 * i + 2] = (_Float16)(v[i].z * inv);
        w[4 * i + 3] = (_Float16)(v[i].w * inv);
      }
    }
    // per-wave staging only: same-wave DS ops complete in order, no barrier needed

    v16h afr[4];
    v8f accR[8], accM[8];
    const v8f vzero = {};

    // GEMM1: rotated = x_hat @ Pi^T
#pragma unroll
    for (int t = 0; t < 4; ++t) afr[t] = load_a_frag(arow0, t, hh);
#pragma unroll
    for (int nt = 0; nt < 8; ++nt) accR[nt] = vzero;
    gemm_16x128(afr, mPiT, lane, accR);

    // quantize to nearest of 4 centroids; stage recon
#pragma unroll
    for (int nt = 0; nt < 8; ++nt) {
#pragma unroll
      for (int j = 0; j < 8; ++j) {
        const float rv = accR[nt][j];
        const float q = (rv < t12) ? ((rv < t01) ? c0 : c1) : ((rv < t23) ? c2 : c3);
        st1[(j + 8 * hh) * PITCH + nt * 16 + ln] = (_Float16)q;
      }
    }

    // GEMM2: x_mse = recon @ Pi (kept in registers)
#pragma unroll
    for (int t = 0; t < 4; ++t) afr[t] = load_a_frag(arow1, t, hh);
#pragma unroll
    for (int nt = 0; nt < 8; ++nt) accM[nt] = vzero;
    gemm_16x128(afr, mPi, lane, accM);

    // residual + residual norm; stage residual
    float pr[8];
#pragma unroll
    for (int j = 0; j < 8; ++j) pr[j] = 0.f;
#pragma unroll
    for (int nt = 0; nt < 8; ++nt) {
#pragma unroll
      for (int j = 0; j < 8; ++j) {
        const int row = j + 8 * hh, col = nt * 16 + ln;
        const float xh = (float)st0[row * PITCH + col];
        const float rs = xh - accM[nt][j];
        pr[j] += rs * rs;
        st1[row * PITCH + col] = (_Float16)rs;
      }
    }
    float resn[8];
#pragma unroll
    for (int j = 0; j < 8; ++j) {  // reduce across the 16-lane half (rows stay local)
      float p = pr[j];
      p += __shfl_xor(p, 1, 32);
      p += __shfl_xor(p, 2, 32);
      p += __shfl_xor(p, 4, 32);
      p += __shfl_xor(p, 8, 32);
      resn[j] = (float)(_Float16)sqrtf(p);  // fp16 round-trip
    }

    // GEMM3: projected = residual @ S^T
#pragma unroll
    for (int t = 0; t < 4; ++t) afr[t] = load_a_frag(arow1, t, hh);
#pragma unroll
    for (int nt = 0; nt < 8; ++nt) accR[nt] = vzero;
    gemm_16x128(afr, mST, lane, accR);

    // signs -> stage
#pragma unroll
    for (int nt = 0; nt < 8; ++nt)
#pragma unroll
      for (int j = 0; j < 8; ++j)
        st1[(j + 8 * hh) * PITCH + nt * 16 + ln] =
            (_Float16)((accR[nt][j] >= 0.f) ? 1.f : -1.f);

    // GEMM4: qjl = signs @ S
#pragma unroll
    for (int t = 0; t < 4; ++t) afr[t] = load_a_frag(arow1, t, hh);
#pragma unroll
    for (int nt = 0; nt < 8; ++nt) accR[nt] = vzero;
    gemm_16x128(afr, mS, lane, accR);

    // reconstruct + store
    float vn[8];
#pragma unroll
    for (int j = 0; j < 8; ++j) vn[j] = snorm[wv * 16 + j + 8 * hh];
#pragma unroll
    for (int nt = 0; nt < 8; ++nt) {
#pragma unroll
      for (int j = 0; j < 8; ++j) {
        const int row = j + 8 * hh, col = nt * 16 + ln;
        out[(size_t)(rowBase + row) * DIM + col] =
            vn[j] * (accM[nt][j] + CORR * resn[j] * accR[nt][j]);
      }
    }
  }
}

extern "C" void kernel_launch(void* const* d_in, const int* in_sizes, int n_in,
                              void* d_out, int out_size, void* d_ws, size_t ws_size,
                              hipStream_t stream) {
  const float* x    = (const float*)d_in[0];
  const float* Pi   = (const float*)d_in[1];
  const float* S    = (const float*)d_in[2];
  const float* cent = (const float*)d_in[3];
  float* out = (float*)d_out;

  const int nrows  = in_sizes[0] / DIM;            // 262144
  const int nTiles = nrows / ROWS_PER_BLOCK;       // 4096
  const int blocks = (nTiles < GRID_BLOCKS) ? nTiles : GRID_BLOCKS;
  const size_t lds = (size_t)(4 * MATH + WAVES * 2 * STAGEH) * sizeof(_Float16)
                   + (size_t)WAVES * 16 * sizeof(float);  // ~170 KB (<= 320 KB/WGP)
  turboquant_roundtrip_kernel<<<blocks, WAVES * 32, lds, stream>>>(x, Pi, S, cent, out,
                                                                   nTiles);
}
